// AceStepLyricEncoder_63788854280668
// MI455X (gfx1250) — matrix operand order repaired and checked by reading, not compiled, not run on hardware
//
#include <hip/hip_runtime.h>
#include <hip/hip_bf16.h>

#define NL    8
#define NH    16
#define NKV   8
#define DHD   128
#define HID   2048
#define INTS  6144
#define TIN   1024
#define WIN   128
#define NB    2
#define SEQ   1024
#define EPSV  1e-6f
#define THETA 1000000.0f

#define BM 128
#define BN 128
#define BK 32

typedef __bf16 bf16_t;
typedef __attribute__((ext_vector_type(16))) __bf16 bf16x16;
typedef __attribute__((ext_vector_type(8)))  float  f32x8;

union Frag32B { bf16x16 v; uint4 q[2]; };

#if __has_builtin(__builtin_amdgcn_global_load_async_to_lds_b128) && \
    __has_builtin(__builtin_amdgcn_s_wait_asynccnt)
#define HAVE_ASYNC_LDS 1
#else
#define HAVE_ASYNC_LDS 0
#endif

#if HAVE_ASYNC_LDS
typedef int v4i_t __attribute__((vector_size(16)));
typedef __attribute__((address_space(1))) v4i_t as1_v4i;
typedef __attribute__((address_space(3))) v4i_t as3_v4i;
static __device__ __forceinline__ void async_cp16(const void* g, void* l) {
    // copy 16 bytes global -> LDS via the ASYNCcnt-tracked path
    __builtin_amdgcn_global_load_async_to_lds_b128(
        (as1_v4i*)(unsigned long long)g,
        (as3_v4i*)(unsigned)(unsigned long long)l, 0, 0);
}
#endif

static __device__ __forceinline__ f32x8 wmma_bf16(bf16x16 a, bf16x16 b, f32x8 c) {
    return __builtin_amdgcn_wmma_f32_16x16x32_bf16(false, a, false, b, (short)0, c, false, false);
}
static __device__ __forceinline__ f32x8 zero8() {
    f32x8 z; for (int i = 0; i < 8; i++) z[i] = 0.f; return z;
}

// ---------------- f32 -> bf16 convert ----------------
__global__ void k_cvt_bf16(const float* __restrict__ src, bf16_t* __restrict__ dst, int n) {
    int i = (blockIdx.x * 256 + threadIdx.x) * 4;
    if (i + 3 < n) {
        float4 v = *(const float4*)(src + i);
        dst[i+0] = (bf16_t)v.x; dst[i+1] = (bf16_t)v.y;
        dst[i+2] = (bf16_t)v.z; dst[i+3] = (bf16_t)v.w;
    }
}

// ---------------- RMSNorm: f32 in, bf16 or f32 out ----------------
__global__ __launch_bounds__(256) void k_rmsnorm(const float* __restrict__ x,
                                                 const float* __restrict__ w,
                                                 bf16_t* __restrict__ outb,
                                                 float* __restrict__ outf, int C) {
    int row = blockIdx.x;
    const float* xr = x + (size_t)row * C;
    int tid = threadIdx.x;
    float ss = 0.f;
    for (int c = tid; c < C; c += 256) { float v = xr[c]; ss += v * v; }
    __shared__ float red[8];
    for (int m = 16; m; m >>= 1) ss += __shfl_xor(ss, m, 32);
    if ((tid & 31) == 0) red[tid >> 5] = ss;
    __syncthreads();
    if (tid < 8) {
        float v = red[tid];
        v += __shfl_xor(v, 4, 32);
        v += __shfl_xor(v, 2, 32);
        v += __shfl_xor(v, 1, 32);
        if (tid == 0) red[0] = v;
    }
    __syncthreads();
    float inv = rsqrtf(red[0] / (float)C + EPSV);
    for (int c = tid; c < C; c += 256) {
        float y = xr[c] * inv * w[c];
        if (outb) outb[(size_t)row * C + c] = (bf16_t)y;
        else      outf[(size_t)row * C + c] = y;
    }
}

// ---- GEMM: C[M,N] = A(bf16)[M,K] * W(f32)[N,K]^T (+R), double-buffered LDS ----
__global__ __launch_bounds__(256) void k_gemm(const bf16_t* __restrict__ A,
                                              const float* __restrict__ W,
                                              const float* __restrict__ R,
                                              float* __restrict__ C,
                                              int M, int N, int K) {
    __shared__ __align__(32) bf16_t sA[2][BM][BK + 16];
    __shared__ __align__(32) bf16_t sW[2][BN][BK + 16];
    const int tid  = threadIdx.x;
    const int lane = tid & 31, wave = tid >> 5;
    const int wm = wave & 3, wn = wave >> 2;        // 4x2 wave grid
    const int m0 = blockIdx.y * BM, n0 = blockIdx.x * BN;
    const int llo = lane & 15, half = lane >> 4;

    f32x8 acc[2][4];
    for (int i = 0; i < 2; i++) for (int j = 0; j < 4; j++) acc[i][j] = zero8();

    const int ldr = tid >> 1;           // 0..127
    const int lds = (tid & 1) * 16;     // 0 or 16
    const bf16_t* arow = A + (size_t)(m0 + ldr) * K + lds;
    const float*  wrow = W + (size_t)(n0 + ldr) * K + lds;

    // ---- prologue: stage k-step 0 into buffer 0 ----
    {
#if HAVE_ASYNC_LDS
        async_cp16(arow,     &sA[0][ldr][lds + 0]);
        async_cp16(arow + 8, &sA[0][ldr][lds + 8]);
#else
        *(uint4*)&sA[0][ldr][lds + 0] = *(const uint4*)(arow);
        *(uint4*)&sA[0][ldr][lds + 8] = *(const uint4*)(arow + 8);
#endif
        float4 w0 = *(const float4*)(wrow + 0);
        float4 w1 = *(const float4*)(wrow + 4);
        float4 w2 = *(const float4*)(wrow + 8);
        float4 w3 = *(const float4*)(wrow + 12);
        union { bf16_t h[16]; uint4 q[2]; } wc;
        float wf[16] = { w0.x,w0.y,w0.z,w0.w, w1.x,w1.y,w1.z,w1.w,
                         w2.x,w2.y,w2.z,w2.w, w3.x,w3.y,w3.z,w3.w };
        for (int t = 0; t < 16; t++) wc.h[t] = (bf16_t)wf[t];
        *(uint4*)&sW[0][ldr][lds + 0] = wc.q[0];
        *(uint4*)&sW[0][ldr][lds + 8] = wc.q[1];
#if HAVE_ASYNC_LDS
        __builtin_amdgcn_s_wait_asynccnt(0);
#endif
    }
    __syncthreads();

    int buf = 0;
    for (int kk = 0; kk < K; kk += BK, buf ^= 1) {
        const int nxt = buf ^ 1;
        const bool hasnext = (kk + BK) < K;
        float4 w0, w1, w2, w3;
        if (hasnext) {
            // issue next-stage loads before computing on the current stage
            const bf16_t* ap = arow + kk + BK;
#if HAVE_ASYNC_LDS
            async_cp16(ap,     &sA[nxt][ldr][lds + 0]);
            async_cp16(ap + 8, &sA[nxt][ldr][lds + 8]);
#else
            *(uint4*)&sA[nxt][ldr][lds + 0] = *(const uint4*)(ap);
            *(uint4*)&sA[nxt][ldr][lds + 8] = *(const uint4*)(ap + 8);
#endif
            const float* wp = wrow + kk + BK;
            w0 = *(const float4*)(wp + 0);
            w1 = *(const float4*)(wp + 4);
            w2 = *(const float4*)(wp + 8);
            w3 = *(const float4*)(wp + 12);
            if (kk + 2 * BK < K) {
                __builtin_prefetch((const void*)(wrow + kk + 2 * BK), 0, 1);
                __builtin_prefetch((const void*)(arow + kk + 2 * BK), 0, 1);
            }
        }
        // fragments from current buffer
        bf16x16 af[2], bfr[4];
        for (int i = 0; i < 2; i++) {
            const bf16_t* pa = &sA[buf][wm * 32 + i * 16 + llo][half * 16];
            Frag32B f;
            f.q[0] = *(const uint4*)pa;
            f.q[1] = *(const uint4*)(pa + 8);
            af[i] = f.v;
        }
        for (int j = 0; j < 4; j++) {
            bfr[j] = *(const bf16x16*)&sW[buf][wn * 64 + j * 16 + llo][half * 16];
        }
        for (int i = 0; i < 2; i++)
            for (int j = 0; j < 4; j++)
                acc[i][j] = wmma_bf16(af[i], bfr[j], acc[i][j]);

        if (hasnext) {
            union { bf16_t h[16]; uint4 q[2]; } wc;
            float wf[16] = { w0.x,w0.y,w0.z,w0.w, w1.x,w1.y,w1.z,w1.w,
                             w2.x,w2.y,w2.z,w2.w, w3.x,w3.y,w3.z,w3.w };
            for (int t = 0; t < 16; t++) wc.h[t] = (bf16_t)wf[t];
            *(uint4*)&sW[nxt][ldr][lds + 0] = wc.q[0];
            *(uint4*)&sW[nxt][ldr][lds + 8] = wc.q[1];
#if HAVE_ASYNC_LDS
            __builtin_amdgcn_s_wait_asynccnt(0);
#endif
        }
        __syncthreads();
    }
    // epilogue: C layout row = r + 8*half, col = llo
    for (int i = 0; i < 2; i++) {
        for (int j = 0; j < 4; j++) {
            for (int r = 0; r < 8; r++) {
                int row = m0 + wm * 32 + i * 16 + r + 8 * half;
                int col = n0 + wn * 64 + j * 16 + llo;
                size_t idx = (size_t)row * N + col;
                float v = acc[i][j][r];
                if (R) v += R[idx];
                C[idx] = v;
            }
        }
    }
}

// ---------------- per-head q/k RMSNorm + RoPE prep, V transpose ----------------
__global__ __launch_bounds__(128) void k_qkprep(const float* __restrict__ qraw,
                                                const float* __restrict__ kraw,
                                                const float* __restrict__ vraw,
                                                const float* __restrict__ qn,
                                                const float* __restrict__ kn,
                                                bf16_t* __restrict__ Qb,
                                                bf16_t* __restrict__ Kb,
                                                bf16_t* __restrict__ Vt) {
    int row = blockIdx.x;               // b*SEQ + s
    int b = row / SEQ, s = row % SEQ;
    int d = threadIdx.x;                // 0..127
    __shared__ float buf[DHD];
    __shared__ float red[4];
    float fr  = (float)(d & 63) * (2.0f / (float)DHD);
    float inv = __powf(THETA, -fr);
    float ang = (float)s * inv;
    float cs, sn;
    __sincosf(ang, &cs, &sn);
    int lane = threadIdx.x & 31, wv = threadIdx.x >> 5;

    for (int h = 0; h < NH; h++) {
        float v = qraw[(size_t)row * (NH * DHD) + h * DHD + d];
        float ss = v * v;
        for (int m = 16; m; m >>= 1) ss += __shfl_xor(ss, m, 32);
        if (lane == 0) red[wv] = ss;
        __syncthreads();
        float tot = red[0] + red[1] + red[2] + red[3];
        float y = v * rsqrtf(tot / (float)DHD + EPSV) * qn[d];
        buf[d] = y;
        __syncthreads();
        float other = (d < 64) ? -buf[d + 64] : buf[d - 64];
        Qb[(((size_t)b * NH + h) * SEQ + s) * DHD + d] = (bf16_t)(y * cs + other * sn);
        __syncthreads();
    }
    for (int h = 0; h < NKV; h++) {
        float v = kraw[(size_t)row * (NKV * DHD) + h * DHD + d];
        float ss = v * v;
        for (int m = 16; m; m >>= 1) ss += __shfl_xor(ss, m, 32);
        if (lane == 0) red[wv] = ss;
        __syncthreads();
        float tot = red[0] + red[1] + red[2] + red[3];
        float y = v * rsqrtf(tot / (float)DHD + EPSV) * kn[d];
        buf[d] = y;
        __syncthreads();
        float other = (d < 64) ? -buf[d + 64] : buf[d - 64];
        Kb[(((size_t)b * NKV + h) * SEQ + s) * DHD + d] = (bf16_t)(y * cs + other * sn);
        float vv = vraw[(size_t)row * (NKV * DHD) + h * DHD + d];
        Vt[(((size_t)b * NKV + h) * DHD + d) * SEQ + s] = (bf16_t)vv;   // transposed [D,S]
        __syncthreads();
    }
}

// ---------------- flash attention (WMMA, online softmax) ----------------
__global__ __launch_bounds__(128) void k_attn(const bf16_t* __restrict__ Q,
                                              const bf16_t* __restrict__ K,
                                              const bf16_t* __restrict__ Vt,
                                              bf16_t* __restrict__ O, int window) {
    __shared__ __align__(32) bf16_t sP[4][16][72];
    const int tid = threadIdx.x, lane = tid & 31, wave = tid >> 5;
    const int llo = lane & 15, half = lane >> 4;
    const int b = blockIdx.z, h = blockIdx.y, hk = h >> 1;
    const int qb = blockIdx.x * 64;
    const int qw = qb + wave * 16;
    const bf16_t* Qh = Q  + ((size_t)b * NH  + h ) * SEQ * DHD;
    const bf16_t* Kh = K  + ((size_t)b * NKV + hk) * SEQ * DHD;
    const bf16_t* Vh = Vt + ((size_t)b * NKV + hk) * DHD * SEQ;

    bf16x16 qf[4];
    for (int dk = 0; dk < 4; dk++) {
        const bf16_t* p = Qh + (size_t)(qw + llo) * DHD + dk * 32 + half * 16;
        Frag32B f;
        f.q[0] = *(const uint4*)p;
        f.q[1] = *(const uint4*)(p + 8);
        qf[dk] = f.v;
    }
    f32x8 out[8];
    for (int dt = 0; dt < 8; dt++) out[dt] = zero8();
    float Mst[8], Lst[8];
    for (int r = 0; r < 8; r++) { Mst[r] = -1e30f; Lst[r] = 0.f; }

    int k0s = 0, k0e = SEQ;
    if (window) {
        k0s = qb - WIN; if (k0s < 0) k0s = 0;
        k0e = qb + 64 + WIN; if (k0e > SEQ) k0e = SEQ;
    }
    const float scale = 0.08838834764831845f;   // 1/sqrt(128)

    for (int k0 = k0s; k0 < k0e; k0 += 64) {
        if (k0 + 64 < k0e) {
            __builtin_prefetch((const void*)(Kh + (size_t)(k0 + 64 + lane * 2) * DHD), 0, 1);
            __builtin_prefetch((const void*)(Vh + (size_t)(lane * 4) * SEQ + k0 + 64), 0, 1);
        }
        f32x8 sacc[4];
        for (int t = 0; t < 4; t++) sacc[t] = zero8();
        for (int t = 0; t < 4; t++) {
            for (int dk = 0; dk < 4; dk++) {
                const bf16_t* p = Kh + (size_t)(k0 + t * 16 + llo) * DHD + dk * 32 + half * 16;
                bf16x16 kf = *(const bf16x16*)p;
                sacc[t] = wmma_bf16(qf[dk], kf, sacc[t]);
            }
        }
        // online softmax in C-fragment layout: row = r + 8*half, col = k0 + t*16 + llo
        for (int r = 0; r < 8; r++) {
            float s[4];
            for (int t = 0; t < 4; t++) s[t] = sacc[t][r] * scale;
            if (window) {
                int qi = qw + r + 8 * half;
                for (int t = 0; t < 4; t++) {
                    int diff = qi - (k0 + t * 16 + llo);
                    if (diff < 0) diff = -diff;
                    if (diff > WIN) s[t] = -1e30f;
                }
            }
            float mr = fmaxf(fmaxf(s[0], s[1]), fmaxf(s[2], s[3]));
            for (int m = 1; m <= 8; m <<= 1) mr = fmaxf(mr, __shfl_xor(mr, m, 32));
            float Mn = fmaxf(Mst[r], mr);
            float sc = __expf(Mst[r] - Mn);
            float rs = 0.f;
            for (int t = 0; t < 4; t++) {
                float p = __expf(s[t] - Mn);
                rs += p;
                sP[wave][r + 8 * half][t * 16 + llo] = (bf16_t)p;
            }
            for (int m = 1; m <= 8; m <<= 1) rs += __shfl_xor(rs, m, 32);
            Lst[r] = Lst[r] * sc + rs;
            Mst[r] = Mn;
            for (int dt = 0; dt < 8; dt++) out[dt][r] = out[dt][r] * sc;
        }
        __syncthreads();
        bf16x16 pf[2];
        for (int kk = 0; kk < 2; kk++) {
            const bf16_t* p = &sP[wave][llo][kk * 32 + half * 16];
            Frag32B f;
            f.q[0] = *(const uint4*)p;
            f.q[1] = *(const uint4*)(p + 8);
            pf[kk] = f.v;
        }
        for (int dt = 0; dt < 8; dt++) {
            for (int kk = 0; kk < 2; kk++) {
                const bf16_t* vp = Vh + (size_t)(dt * 16 + llo) * SEQ + k0 + kk * 32 + half * 16;
                bf16x16 vf = *(const bf16x16*)vp;
                out[dt] = wmma_bf16(pf[kk], vf, out[dt]);
            }
        }
        __syncthreads();
    }
    // write O as [B,S,H*D] bf16 (A-matrix for the o-projection GEMM)
    for (int dt = 0; dt < 8; dt++) {
        for (int r = 0; r < 8; r++) {
            int q = qw + r + 8 * half;
            int d = dt * 16 + llo;
            float v = out[dt][r] / Lst[r];
            O[(((size_t)b * SEQ + q) * NH + h) * DHD + d] = (bf16_t)v;
        }
    }
}

// ---------------- SwiGLU: a = silu(g) * u -> bf16 ----------------
__global__ void k_swiglu(const float* __restrict__ g, const float* __restrict__ u,
                         bf16_t* __restrict__ out, int n) {
    int i = blockIdx.x * 256 + threadIdx.x;
    if (i < n) {
        float gv = g[i];
        float s = gv / (1.f + __expf(-gv));
        out[i] = (bf16_t)(s * u[i]);
    }
}

extern "C" void kernel_launch(void* const* d_in, const int* in_sizes, int n_in,
                              void* d_out, int out_size, void* d_ws, size_t ws_size,
                              hipStream_t stream) {
    const float* x      = (const float*)d_in[0];
    const float* proj_w = (const float*)d_in[1];
    const float* Wq     = (const float*)d_in[2];
    const float* Wk     = (const float*)d_in[3];
    const float* Wv     = (const float*)d_in[4];
    const float* Wo     = (const float*)d_in[5];
    const float* qn     = (const float*)d_in[6];
    const float* kn     = (const float*)d_in[7];
    const float* ln1    = (const float*)d_in[8];
    const float* ln2    = (const float*)d_in[9];
    const float* Wg     = (const float*)d_in[10];
    const float* Wu     = (const float*)d_in[11];
    const float* Wd     = (const float*)d_in[12];
    const float* normw  = (const float*)d_in[13];

    const int M = NB * SEQ;   // 2048 token rows
    size_t off = 0;
    char* base = (char*)d_ws;
    auto alloc = [&](size_t bytes) -> void* {
        off = (off + 255) & ~(size_t)255;
        void* p = base + off; off += bytes; return p;
    };
    bf16_t* xb   = (bf16_t*)alloc((size_t)M * TIN  * 2);
    float*  hbuf = (float*) alloc((size_t)M * HID  * 4);
    bf16_t* hnb  = (bf16_t*)alloc((size_t)M * HID  * 2);
    float*  qraw = (float*) alloc((size_t)M * NH  * DHD * 4);
    float*  kraw = (float*) alloc((size_t)M * NKV * DHD * 4);
    float*  vraw = (float*) alloc((size_t)M * NKV * DHD * 4);
    bf16_t* qbf  = (bf16_t*)alloc((size_t)M * NH  * DHD * 2);
    bf16_t* kbf  = (bf16_t*)alloc((size_t)M * NKV * DHD * 2);
    bf16_t* vtb  = (bf16_t*)alloc((size_t)M * NKV * DHD * 2);
    bf16_t* obf  = (bf16_t*)alloc((size_t)M * NH  * DHD * 2);
    float*  gbuf = (float*) alloc((size_t)M * INTS * 4);
    float*  ubuf = (float*) alloc((size_t)M * INTS * 4);
    bf16_t* abf  = (bf16_t*)alloc((size_t)M * INTS * 2);

    dim3 blk256(256), blk128(128);

    k_cvt_bf16<<<dim3((M * TIN / 4 + 255) / 256), blk256, 0, stream>>>(x, xb, M * TIN);
    k_gemm<<<dim3(HID / BN, M / BM), blk256, 0, stream>>>(xb, proj_w, nullptr, hbuf, M, HID, TIN);

    for (int i = 0; i < NL; i++) {
        k_rmsnorm<<<dim3(M), blk256, 0, stream>>>(hbuf, ln1 + (size_t)i * HID, hnb, nullptr, HID);
        k_gemm<<<dim3((NH * DHD) / BN, M / BM), blk256, 0, stream>>>(
            hnb, Wq + (size_t)i * NH * DHD * HID, nullptr, qraw, M, NH * DHD, HID);
        k_gemm<<<dim3((NKV * DHD) / BN, M / BM), blk256, 0, stream>>>(
            hnb, Wk + (size_t)i * NKV * DHD * HID, nullptr, kraw, M, NKV * DHD, HID);
        k_gemm<<<dim3((NKV * DHD) / BN, M / BM), blk256, 0, stream>>>(
            hnb, Wv + (size_t)i * NKV * DHD * HID, nullptr, vraw, M, NKV * DHD, HID);
        k_qkprep<<<dim3(M), blk128, 0, stream>>>(qraw, kraw, vraw,
            qn + (size_t)i * DHD, kn + (size_t)i * DHD, qbf, kbf, vtb);
        k_attn<<<dim3(SEQ / 64, NH, NB), blk128, 0, stream>>>(qbf, kbf, vtb, obf,
            ((i & 1) == 0) ? 1 : 0);
        k_gemm<<<dim3(HID / BN, M / BM), blk256, 0, stream>>>(
            obf, Wo + (size_t)i * HID * NH * DHD, hbuf, hbuf, M, HID, NH * DHD);
        k_rmsnorm<<<dim3(M), blk256, 0, stream>>>(hbuf, ln2 + (size_t)i * HID, hnb, nullptr, HID);
        k_gemm<<<dim3(INTS / BN, M / BM), blk256, 0, stream>>>(
            hnb, Wg + (size_t)i * INTS * HID, nullptr, gbuf, M, INTS, HID);
        k_gemm<<<dim3(INTS / BN, M / BM), blk256, 0, stream>>>(
            hnb, Wu + (size_t)i * INTS * HID, nullptr, ubuf, M, INTS, HID);
        k_swiglu<<<dim3((M * INTS + 255) / 256), blk256, 0, stream>>>(gbuf, ubuf, abf, M * INTS);
        k_gemm<<<dim3(HID / BN, M / BM), blk256, 0, stream>>>(
            abf, Wd + (size_t)i * HID * INTS, hbuf, hbuf, M, HID, INTS);
    }
    k_rmsnorm<<<dim3(M), blk256, 0, stream>>>(hbuf, normw, nullptr, (float*)d_out, HID);
}
